// EdgeConvBlock_6219112644823
// MI455X (gfx1250) — matrix operand order repaired and checked by reading, not compiled, not run on hardware
//
#include <hip/hip_runtime.h>
#include <hip/hip_bf16.h>
#include <cstdint>
#include <cstddef>

typedef __attribute__((ext_vector_type(16))) __bf16 v16bf;
typedef __attribute__((ext_vector_type(8)))  __bf16 v8bf;
typedef __attribute__((ext_vector_type(8)))  float  v8f;

#define BN_EPS 1e-5f

// ------------------------------------------------------------------
// kNN: one block (256 threads) per query point, 17 iterative arg-min
// passes over per-thread register-resident distances (2048 candidates).
// ------------------------------------------------------------------
__global__ __launch_bounds__(256) void knn_kernel(const float* __restrict__ points,
                                                  int* __restrict__ idx)
{
    const int P = 2048;
    int pt = blockIdx.x;              // 0 .. N*P-1
    int n  = pt >> 11;
    int p  = pt & (P - 1);
    const float* pts = points + (size_t)n * P * 3;
    float px = pts[p * 3 + 0], py = pts[p * 3 + 1], pz = pts[p * 3 + 2];

    float d[8]; int qi[8];
#pragma unroll
    for (int j = 0; j < 8; ++j) {
        int q = threadIdx.x + j * 256;
        float dx = pts[q * 3 + 0] - px;
        float dy = pts[q * 3 + 1] - py;
        float dz = pts[q * 3 + 2] - pz;
        d[j]  = dx * dx + dy * dy + dz * dz;
        qi[j] = q;
    }

    __shared__ unsigned long long red[256];
    unsigned long long prev = 0ull;   // keys are >= 1 (index biased by +1)
    for (int s = 0; s < 17; ++s) {
        unsigned long long best = ~0ull;
#pragma unroll
        for (int j = 0; j < 8; ++j) {
            unsigned long long key =
                ((unsigned long long)__float_as_uint(d[j]) << 32) | (unsigned)(qi[j] + 1);
            if (key > prev && key < best) best = key;
        }
        red[threadIdx.x] = best;
        __syncthreads();
        for (int off = 128; off > 0; off >>= 1) {
            if ((int)threadIdx.x < off) {
                unsigned long long o = red[threadIdx.x + off];
                if (o < red[threadIdx.x]) red[threadIdx.x] = o;
            }
            __syncthreads();
        }
        unsigned long long winner = red[0];
        __syncthreads();
        if (s >= 1 && threadIdx.x == 0)
            idx[pt * 16 + (s - 1)] = (int)(winner & 0xffffffffu) - 1;
        prev = winner;
    }
}

// ------------------------------------------------------------------
// Edge features: edge[row, 0:64] = center, edge[row, 64:128] = nbr-center (bf16)
// ------------------------------------------------------------------
__global__ __launch_bounds__(256) void edge_kernel(const float* __restrict__ feat,
                                                   const int* __restrict__ idx,
                                                   __bf16* __restrict__ edge, int total)
{
    int i = blockIdx.x * 256 + threadIdx.x;
    if (i >= total) return;
    int c   = i & 63;
    int row = i >> 6;                 // pt*16 + k
    int pt  = row >> 4;
    int n   = pt >> 11;
    int nb  = idx[row];
    float cf = feat[(size_t)pt * 64 + c];
    float nf = feat[(size_t)(n * 2048 + nb) * 64 + c];
    __bf16* e = edge + (size_t)row * 128;
    e[c]      = (__bf16)cf;
    e[64 + c] = (__bf16)(nf - cf);
}

__global__ __launch_bounds__(256) void cvt_bf16_kernel(const float* __restrict__ src,
                                                       __bf16* __restrict__ dst, int n)
{
    int i = blockIdx.x * 256 + threadIdx.x;
    if (i < n) dst[i] = (__bf16)src[i];
}

// ------------------------------------------------------------------
// Attention head scores: per point, WMMA 16x32 (edge 16x128 @ w1^T) ->
// leaky-relu -> 32->4 projection via wave shuffles -> softmax over K.
// ------------------------------------------------------------------
__global__ __launch_bounds__(256) void attn_kernel(
    const __bf16* __restrict__ edge, const __bf16* __restrict__ w1,
    const float* __restrict__ b1, const float* __restrict__ w2,
    const float* __restrict__ b2, const float* __restrict__ tau,
    float* __restrict__ attw)
{
    __shared__ float amat[8][16][4];
    const int lane  = threadIdx.x & 31;
    const int half  = lane >> 4;
    const int row   = lane & 15;
    const int wslot = threadIdx.x >> 5;
    const int pt    = blockIdx.x * 8 + wslot;

    v8f acc0 = {}, acc1 = {};
    const __bf16* abase = edge + (size_t)pt * 16 * 128;
#pragma unroll
    for (int ks = 0; ks < 4; ++ks) {
        union { v16bf v; v8bf h[2]; } au;
        const __bf16* ap = abase + (size_t)row * 128 + ks * 32 + half * 16;
        au.h[0] = *(const v8bf*)(ap);
        au.h[1] = *(const v8bf*)(ap + 8);
        union { v16bf v; v8bf h[2]; } b0u, b1u;
        const __bf16* bp0 = w1 + (size_t)row * 128 + ks * 32 + half * 16;
        const __bf16* bp1 = w1 + (size_t)(16 + row) * 128 + ks * 32 + half * 16;
        b0u.h[0] = *(const v8bf*)(bp0); b0u.h[1] = *(const v8bf*)(bp0 + 8);
        b1u.h[0] = *(const v8bf*)(bp1); b1u.h[1] = *(const v8bf*)(bp1 + 8);
        acc0 = __builtin_amdgcn_wmma_f32_16x16x32_bf16(false, au.v, false, b0u.v,
                                                       (short)0, acc0, false, false);
        acc1 = __builtin_amdgcn_wmma_f32_16x16x32_bf16(false, au.v, false, b1u.v,
                                                       (short)0, acc1, false, false);
    }
    const int hid0 = row, hid1 = 16 + row;
    const float bb0 = b1[hid0], bb1 = b1[hid1];
    float h0[8], h1[8];
#pragma unroll
    for (int r = 0; r < 8; ++r) {
        float v0 = acc0[r] + bb0; h0[r] = v0 > 0.f ? v0 : 0.2f * v0;
        float v1 = acc1[r] + bb1; h1[r] = v1 > 0.f ? v1 : 0.2f * v1;
    }
#pragma unroll
    for (int r = 0; r < 8; ++r) {
#pragma unroll
        for (int o = 0; o < 4; ++o) {
            float part = h0[r] * w2[o * 32 + hid0] + h1[r] * w2[o * 32 + hid1];
            part += __shfl_xor(part, 1, 32);
            part += __shfl_xor(part, 2, 32);
            part += __shfl_xor(part, 4, 32);
            part += __shfl_xor(part, 8, 32);
            if (row == 0) amat[wslot][r + half * 8][o] = part + b2[o];
        }
    }
    __syncthreads();
    if (lane < 4) {
        const float itau = 1.0f / tau[0];
        const int o = lane;
        float vals[16], mx = -3.0e38f;
#pragma unroll
        for (int k = 0; k < 16; ++k) {
            vals[k] = amat[wslot][k][o] * itau;
            mx = fmaxf(mx, vals[k]);
        }
        float sum = 0.f;
#pragma unroll
        for (int k = 0; k < 16; ++k) { vals[k] = expf(vals[k] - mx); sum += vals[k]; }
        float inv = 1.0f / sum;
#pragma unroll
        for (int k = 0; k < 16; ++k)
            attw[((size_t)pt * 16 + k) * 4 + o] = vals[k] * inv;
    }
}

// ------------------------------------------------------------------
// bf16 GEMM: Y[M x NOUT] = A[M x KD] @ W^T, W is (NOUT x KD) row-major.
// W is staged once per workgroup into LDS with ASYNCcnt-tracked
// global_load_async_to_lds_b128; B-fragments then come from LDS
// (ds_load_b128), A-fragments stream from global (b128 clauses).
// Per-channel sum/sumsq for BN via LDS reduce + one atomic per block.
// ------------------------------------------------------------------
template<int KD, int NT>
__global__ __launch_bounds__(256) void gemm_bf16_kernel(
    const __bf16* __restrict__ A, const __bf16* __restrict__ W,
    __bf16* __restrict__ Y, float* __restrict__ gSum, float* __restrict__ gSq,
    int Mrows)
{
    constexpr int NOUT   = NT * 16;
    constexpr int WBYTES = NOUT * KD * 2;          // 8KB or 16KB
    __shared__ __bf16 Wlds[NOUT * KD];
    __shared__ float ls[NOUT];
    __shared__ float lq[NOUT];
    for (int i = threadIdx.x; i < NOUT; i += 256) { ls[i] = 0.f; lq[i] = 0.f; }

    // Cooperative async stage of the weight matrix into LDS, 16B per lane
    // per transfer (ASYNCcnt-tracked; completes before the barrier).
    {
        unsigned ldsbase = (unsigned)(size_t)(void*)&Wlds[0];
#pragma unroll
        for (int bo = 0; bo < WBYTES; bo += 256 * 16) {
            unsigned laddr = ldsbase + bo + threadIdx.x * 16;
            const char* gaddr = (const char*)W + bo + threadIdx.x * 16;
            asm volatile("global_load_async_to_lds_b128 %0, %1, off"
                         :: "v"(laddr), "v"(gaddr) : "memory");
        }
        asm volatile("s_wait_asynccnt 0x0" ::: "memory");
    }
    __syncthreads();

    const int lane = threadIdx.x & 31;
    const int half = lane >> 4;
    const int row  = lane & 15;
    const int wid  = blockIdx.x * 8 + (threadIdx.x >> 5);
    const int m0   = wid * 16;

    if (m0 < Mrows) {                 // wave-uniform (EXEC stays all-ones for WMMA)
        v8f acc[NT] = {};
#pragma unroll
        for (int ks = 0; ks < KD / 32; ++ks) {
            union { v16bf v; v8bf h[2]; } au;
            const __bf16* ap = A + (size_t)(m0 + row) * KD + ks * 32 + half * 16;
            au.h[0] = *(const v8bf*)(ap);
            au.h[1] = *(const v8bf*)(ap + 8);
#pragma unroll
            for (int t = 0; t < NT; ++t) {
                union { v16bf v; v8bf h[2]; } bu;
                const __bf16* bp = &Wlds[(t * 16 + row) * KD + ks * 32 + half * 16];
                bu.h[0] = *(const v8bf*)(bp);
                bu.h[1] = *(const v8bf*)(bp + 8);
                acc[t] = __builtin_amdgcn_wmma_f32_16x16x32_bf16(
                    false, au.v, false, bu.v, (short)0, acc[t], false, false);
            }
        }
#pragma unroll
        for (int t = 0; t < NT; ++t) {
            const int col = t * 16 + row;
            float ps = 0.f, pq = 0.f;
#pragma unroll
            for (int r = 0; r < 8; ++r) {
                float v = acc[t][r];
                int m = m0 + r + half * 8;
                Y[(size_t)m * NOUT + col] = (__bf16)v;
                ps += v; pq += v * v;
            }
            atomicAdd(&ls[col], ps);
            atomicAdd(&lq[col], pq);
        }
    }
    __syncthreads();
    for (int i = threadIdx.x; i < NOUT; i += 256) {
        atomicAdd(&gSum[i], ls[i]);
        atomicAdd(&gSq[i],  lq[i]);
    }
}

// ------------------------------------------------------------------
// BN (global stats) + ReLU, bf16 -> bf16
// ------------------------------------------------------------------
__global__ __launch_bounds__(256) void bnrelu_kernel(
    const __bf16* __restrict__ y, const float* __restrict__ sum,
    const float* __restrict__ sq, const float* __restrict__ g,
    const float* __restrict__ b, __bf16* __restrict__ act,
    int cmask, float invCnt, int total)
{
    int i = blockIdx.x * 256 + threadIdx.x;
    if (i >= total) return;
    int c = i & cmask;
    float mean  = sum[c] * invCnt;
    float var   = sq[c] * invCnt - mean * mean;
    float scale = g[c] * rsqrtf(var + BN_EPS);
    float shift = b[c] - mean * scale;
    float v = (float)y[i] * scale + shift;
    act[i] = (__bf16)fmaxf(v, 0.f);
}

// ------------------------------------------------------------------
// BN+ReLU on conv2 output fused with attention-weighted pooling over K.
// One block (128 threads = channels) per point.
// ------------------------------------------------------------------
__global__ __launch_bounds__(128) void pool_kernel(
    const __bf16* __restrict__ y2, const float* __restrict__ attw,
    const float* __restrict__ s2, const float* __restrict__ q2,
    const float* __restrict__ g2, const float* __restrict__ b2,
    float* __restrict__ xout, float invCnt)
{
    int pt = blockIdx.x;
    int c  = threadIdx.x;
    float mean  = s2[c] * invCnt;
    float var   = q2[c] * invCnt - mean * mean;
    float scale = g2[c] * rsqrtf(var + BN_EPS);
    float shift = b2[c] - mean * scale;
    int h = c >> 5;
    float acc = 0.f;
#pragma unroll
    for (int k = 0; k < 16; ++k) {
        float v = (float)y2[((size_t)pt * 16 + k) * 128 + c] * scale + shift;
        v = fmaxf(v, 0.f);
        acc += v * attw[((size_t)pt * 16 + k) * 4 + h];
    }
    xout[(size_t)pt * 128 + c] = acc;
}

// Per-(n,c) mean over P for SE squeeze
__global__ __launch_bounds__(256) void xmean_kernel(const float* __restrict__ xbuf,
                                                    float* __restrict__ xsum)
{
    int i = blockIdx.x * 256 + threadIdx.x;
    if (i >= 8 * 128) return;
    int n = i >> 7, c = i & 127;
    const float* p = xbuf + (size_t)n * 2048 * 128 + c;
    float s = 0.f;
    for (int q = 0; q < 2048; ++q) s += p[(size_t)q * 128];
    xsum[i] = s;
}

// SE excitation: one wave per batch element
__global__ __launch_bounds__(32) void se_kernel(const float* __restrict__ xsum,
                                                const float* __restrict__ w1,
                                                const float* __restrict__ w2,
                                                float* __restrict__ sesig)
{
    __shared__ float m[128];
    __shared__ float t[32];
    int n = blockIdx.x, lane = threadIdx.x;
    const float invP = 1.0f / 2048.0f;
#pragma unroll
    for (int j = 0; j < 4; ++j) {
        int c = lane + 32 * j;
        m[c] = xsum[n * 128 + c] * invP;
    }
    __syncthreads();
    float a = 0.f;
    for (int c = 0; c < 128; ++c) a += m[c] * w1[lane * 128 + c];
    t[lane] = fmaxf(a, 0.f);
    __syncthreads();
#pragma unroll
    for (int j = 0; j < 4; ++j) {
        int c = lane + 32 * j;
        float s = 0.f;
        for (int h = 0; h < 32; ++h) s += t[h] * w2[c * 32 + h];
        sesig[n * 128 + c] = 1.0f / (1.0f + expf(-s));
    }
}

// out = x * se + relu(bn(shortcut))
__global__ __launch_bounds__(256) void final_kernel(
    const float* __restrict__ xbuf, const float* __restrict__ sesig,
    const __bf16* __restrict__ scb, const float* __restrict__ ssc,
    const float* __restrict__ qsc, const float* __restrict__ g,
    const float* __restrict__ b, float* __restrict__ out,
    float invCnt, int total)
{
    int i = blockIdx.x * 256 + threadIdx.x;
    if (i >= total) return;
    int c  = i & 127;
    int pt = i >> 7;
    int n  = pt >> 11;
    float mean  = ssc[c] * invCnt;
    float var   = qsc[c] * invCnt - mean * mean;
    float scale = g[c] * rsqrtf(var + BN_EPS);
    float shift = b[c] - mean * scale;
    float sc = fmaxf((float)scb[i] * scale + shift, 0.f);
    out[i] = xbuf[i] * sesig[n * 128 + c] + sc;
}

// ------------------------------------------------------------------
extern "C" void kernel_launch(void* const* d_in, const int* in_sizes, int n_in,
                              void* d_out, int out_size, void* d_ws, size_t ws_size,
                              hipStream_t stream)
{
    (void)in_sizes; (void)n_in; (void)out_size; (void)ws_size;
    const float* points   = (const float*)d_in[0];
    const float* features = (const float*)d_in[1];
    const float* conv_w0  = (const float*)d_in[2];
    const float* conv_w1  = (const float*)d_in[3];
    const float* conv_w2  = (const float*)d_in[4];
    const float* bn_g0    = (const float*)d_in[5];
    const float* bn_b0    = (const float*)d_in[6];
    const float* bn_g1    = (const float*)d_in[7];
    const float* bn_b1    = (const float*)d_in[8];
    const float* bn_g2    = (const float*)d_in[9];
    const float* bn_b2    = (const float*)d_in[10];
    const float* attn_w1  = (const float*)d_in[11];
    const float* attn_b1  = (const float*)d_in[12];
    const float* attn_w2  = (const float*)d_in[13];
    const float* attn_b2  = (const float*)d_in[14];
    const float* tau      = (const float*)d_in[15];
    const float* sc_w     = (const float*)d_in[16];
    const float* sc_g     = (const float*)d_in[17];
    const float* sc_b     = (const float*)d_in[18];
    const float* se_w1    = (const float*)d_in[19];
    const float* se_w2    = (const float*)d_in[20];
    float* out = (float*)d_out;

    const int N = 8, P = 2048, PTS = N * P;     // 16384
    const int M = PTS * 16;                     // 262144 edge rows

    char* base = (char*)d_ws;
    size_t off = 0;
    auto take = [&](size_t bytes) -> char* {
        char* p = base + off;
        off = (off + bytes + 255) & ~(size_t)255;
        return p;
    };
    int*    idxb  = (int*)   take((size_t)M * 4);
    __bf16* edgeb = (__bf16*)take((size_t)M * 128 * 2);   // reused as y2
    float*  attwb = (float*) take((size_t)M * 4 * 4);
    __bf16* y01b  = (__bf16*)take((size_t)M * 64 * 2);    // y0, then y1
    __bf16* actb  = (__bf16*)take((size_t)M * 64 * 2);    // act0, then act1
    float*  xbufb = (float*) take((size_t)PTS * 128 * 4);
    __bf16* featb = (__bf16*)take((size_t)PTS * 64 * 2);
    __bf16* scbb  = (__bf16*)take((size_t)PTS * 128 * 2);
    __bf16* wb0   = (__bf16*)take(64 * 128 * 2);
    __bf16* wb1   = (__bf16*)take(64 * 64 * 2);
    __bf16* wb2   = (__bf16*)take(128 * 64 * 2);
    __bf16* wscb  = (__bf16*)take(128 * 64 * 2);
    __bf16* wa1b  = (__bf16*)take(32 * 128 * 2);
    float*  st    = (float*) take(2816 * 4);
    float *s0 = st, *q0 = st + 64, *s1 = st + 128, *q1 = st + 192;
    float *s2 = st + 256, *q2 = st + 384, *ssc = st + 512, *qsc = st + 640;
    float *xsum = st + 768, *sesig = st + 1792;

    hipMemsetAsync(st, 0, 2816 * 4, stream);

    // 1) kNN indices
    knn_kernel<<<PTS, 256, 0, stream>>>(points, idxb);

    // 2) edge features (bf16) + weight/feature conversions
    edge_kernel<<<(M * 64 + 255) / 256, 256, 0, stream>>>(features, idxb, edgeb, M * 64);
    cvt_bf16_kernel<<<(PTS * 64 + 255) / 256, 256, 0, stream>>>(features, featb, PTS * 64);
    cvt_bf16_kernel<<<(64 * 128 + 255) / 256, 256, 0, stream>>>(conv_w0, wb0, 64 * 128);
    cvt_bf16_kernel<<<(64 * 64 + 255) / 256, 256, 0, stream>>>(conv_w1, wb1, 64 * 64);
    cvt_bf16_kernel<<<(128 * 64 + 255) / 256, 256, 0, stream>>>(conv_w2, wb2, 128 * 64);
    cvt_bf16_kernel<<<(128 * 64 + 255) / 256, 256, 0, stream>>>(sc_w, wscb, 128 * 64);
    cvt_bf16_kernel<<<(32 * 128 + 255) / 256, 256, 0, stream>>>(attn_w1, wa1b, 32 * 128);

    // 3) attention weights (WMMA layer-1 + shuffle layer-2 + softmax)
    attn_kernel<<<PTS / 8, 256, 0, stream>>>(edgeb, wa1b, attn_b1, attn_w2, attn_b2,
                                             tau, attwb);

    const float invM   = 1.0f / (float)M;
    const float invPts = 1.0f / (float)PTS;

    // 4) conv stack: GEMM(+stats) -> BN+ReLU, bf16 throughout
    gemm_bf16_kernel<128, 4><<<M / 128, 256, 0, stream>>>(edgeb, wb0, y01b, s0, q0, M);
    bnrelu_kernel<<<(M * 64 + 255) / 256, 256, 0, stream>>>(y01b, s0, q0, bn_g0, bn_b0,
                                                            actb, 63, invM, M * 64);
    gemm_bf16_kernel<64, 4><<<M / 128, 256, 0, stream>>>(actb, wb1, y01b, s1, q1, M);
    bnrelu_kernel<<<(M * 64 + 255) / 256, 256, 0, stream>>>(y01b, s1, q1, bn_g1, bn_b1,
                                                            actb, 63, invM, M * 64);
    gemm_bf16_kernel<64, 8><<<M / 128, 256, 0, stream>>>(actb, wb2, edgeb /*y2*/, s2, q2, M);

    // 5) BN+ReLU fused with attention-weighted K-pooling
    pool_kernel<<<PTS, 128, 0, stream>>>(edgeb, attwb, s2, q2, bn_g2, bn_b2, xbufb, invM);

    // 6) SE block
    xmean_kernel<<<4, 256, 0, stream>>>(xbufb, xsum);
    se_kernel<<<N, 32, 0, stream>>>(xsum, se_w1, se_w2, sesig);

    // 7) shortcut GEMM (+stats), then fused epilogue
    gemm_bf16_kernel<64, 8><<<PTS / 128, 256, 0, stream>>>(featb, wscb, scbb, ssc, qsc, PTS);
    final_kernel<<<(PTS * 128 + 255) / 256, 256, 0, stream>>>(xbufb, sesig, scbb, ssc, qsc,
                                                              sc_g, sc_b, out, invPts,
                                                              PTS * 128);
}
